// AddetiveAttention_42451456754063
// MI455X (gfx1250) — matrix operand order repaired and checked by reading, not compiled, not run on hardware
//
#include <hip/hip_runtime.h>

// AdditiveAttention on gfx1250 (MI455X, wave32, WMMA f16->f32 accumulate).
// B=4, Lq=Lk=512, D=H=128 (fixed per reference).

typedef _Float16 half8 __attribute__((ext_vector_type(8)));
typedef _Float16 v16h  __attribute__((ext_vector_type(16)));
typedef float    v8f   __attribute__((ext_vector_type(8)));
typedef int      v4i_t __attribute__((ext_vector_type(4)));

union V16 { v16h v; half8 h[2]; };

#define MASK_VALUE (-1.0e6f)
#define NEG_BIG    (-3.0e38f)

#if defined(__has_builtin)
#if __has_builtin(__builtin_amdgcn_tanhf)
#define HAVE_TANH_INST 1
#endif
#if __has_builtin(__builtin_amdgcn_global_load_async_to_lds_b128)
#define HAVE_ASYNC_LDS 1
#endif
#endif

__device__ __forceinline__ float fast_tanh(float x) {
#ifdef HAVE_TANH_INST
  return __builtin_amdgcn_tanhf(x);            // native v_tanh_f32 on gfx1250
#else
  float ax = __builtin_fabsf(x);
  float e  = __expf(-2.0f * ax);
  float t  = (1.0f - e) * __builtin_amdgcn_rcpf(1.0f + e);
  return __builtin_copysignf(t, x);
#endif
}

__device__ __forceinline__ void wait_async_lds() {
#if defined(__has_builtin) && __has_builtin(__builtin_amdgcn_s_wait_asynccnt)
  __builtin_amdgcn_s_wait_asynccnt(0);
#elif defined(HAVE_ASYNC_LDS)
  asm volatile("s_wait_asynccnt 0x0" ::: "memory");
#endif
}

// Stage one 16x128 f32 chunk (8192 B) into LDS: 256 threads x 2 x b128.
__device__ __forceinline__ void stage_chunk(const float* __restrict__ g,
                                            float* l, int tid) {
#ifdef HAVE_ASYNC_LDS
  __builtin_amdgcn_global_load_async_to_lds_b128(
      (__attribute__((address_space(1))) v4i_t*)(g + tid * 4),
      (__attribute__((address_space(3))) v4i_t*)(l + tid * 4), 0, 0);
  __builtin_amdgcn_global_load_async_to_lds_b128(
      (__attribute__((address_space(1))) v4i_t*)(g + 1024 + tid * 4),
      (__attribute__((address_space(3))) v4i_t*)(l + 1024 + tid * 4), 0, 0);
#else
  for (int i = tid; i < 2048; i += 256) l[i] = g[i];
#endif
}

// ---------------------------------------------------------------- convert
__global__ void conv_f32_to_f16(const float* __restrict__ src,
                                _Float16* __restrict__ dst, int n) {
  int i = blockIdx.x * blockDim.x + threadIdx.x;
  if (i < n) dst[i] = (_Float16)src[i];
}

// ------------------------------------------------- X(f16 rows x128) @ W(f16 128x128) -> P(f32)
// One block = 16-row tile, 8 waves; wave w owns N-tile [16w,16w+16).
__global__ void proj_gemm(const _Float16* __restrict__ X,
                          const _Float16* __restrict__ W,
                          float* __restrict__ P) {
  const int rt   = blockIdx.x;
  const int tid  = threadIdx.x;
  const int wave = tid >> 5;
  const int lane = tid & 31;
  const bool hi  = lane >= 16;
  const int n0   = wave * 16;
  const int mrow = rt * 16 + (lane & 15);

  const _Float16* xrow = X + mrow * 128;
  v8f c = {};
#pragma unroll
  for (int kc = 0; kc < 4; ++kc) {
    const int k0  = kc * 32;
    const int kLo = k0 + (hi ? 8 : 0);
    const int kHi = k0 + 16 + (hi ? 8 : 0);
    V16 a;
    a.h[0] = *(const half8*)(xrow + kLo);
    a.h[1] = *(const half8*)(xrow + kHi);
    const _Float16* wrow = W + kLo * 128 + n0 + (lane & 15);
    V16 bb;
#pragma unroll
    for (int s = 0; s < 8; ++s) {
      bb.v[s]     = wrow[s * 128];
      bb.v[8 + s] = wrow[(16 + s) * 128];
    }
    c = __builtin_amdgcn_wmma_f32_16x16x32_f16(false, a.v, false, bb.v,
                                               (short)0, c, false, false);
  }
#pragma unroll
  for (int r = 0; r < 8; ++r) {
    const int row = rt * 16 + (hi ? 8 + r : r);
    P[row * 128 + n0 + (lane & 15)] = c[r];
  }
}

// ---------------------------------------------------------------- attention
// Block: (qtile, b). 256 threads = 8 waves.
__global__ void attn_kernel(const float* __restrict__ Qp,
                            const float* __restrict__ Kp,
                            const _Float16* __restrict__ Vh,
                            const _Float16* __restrict__ Woh,
                            const float* __restrict__ wvec,
                            const int* __restrict__ vsl,
                            float* __restrict__ out) {
  const int qt   = blockIdx.x;   // 0..31
  const int b    = blockIdx.y;   // 0..3
  const int tid  = threadIdx.x;  // 0..255
  const int wave = tid >> 5;
  const int lane = tid & 31;
  const bool hi  = lane >= 16;

  __shared__ __attribute__((aligned(16))) float    qp_lds[16 * 128];
  __shared__ __attribute__((aligned(16))) float    kp_lds[2][16 * 128];
  __shared__ __attribute__((aligned(16))) float    scores[16 * 512];
  __shared__ __attribute__((aligned(16))) _Float16 attn_h[16 * 512];
  __shared__ __attribute__((aligned(16))) _Float16 o_lds[16 * 128];
  __shared__ float wv_lds[128];
  __shared__ float red[16 * 16];
  __shared__ float rowstat[16];

  const int vlen = vsl[b];
  const float* KpB = Kp + (size_t)(b * 512) * 128;

  // prefetch Kp chunk 0 (async -> LDS), stage Qp tile + wv (sync)
  stage_chunk(KpB, kp_lds[0], tid);
  for (int i = tid; i < 16 * 128; i += 256)
    qp_lds[i] = Qp[(b * 512 + qt * 16) * 128 + i];
  if (tid < 128) wv_lds[tid] = wvec[tid];
  wait_async_lds();
  __syncthreads();

  // scores[16 x 512] = wv . tanh(q + k)  (the real bottleneck: TRANS/VALU)
  const int q  = tid >> 4;
  const int kk = tid & 15;
  for (int jc = 0; jc < 32; ++jc) {
    const float* cur = kp_lds[jc & 1];
    if (jc + 1 < 32)                               // overlap next chunk with compute
      stage_chunk(KpB + (jc + 1) * 16 * 128, kp_lds[(jc + 1) & 1], tid);
    const int j = jc * 16 + kk;
    float acc = 0.0f;
#pragma unroll 4
    for (int h = 0; h < 128; ++h)
      acc += wv_lds[h] * fast_tanh(qp_lds[q * 128 + h] + cur[kk * 128 + h]);
    scores[q * 512 + j] = (j < vlen) ? acc : MASK_VALUE;
    wait_async_lds();
    __syncthreads();
  }

  // two-pass softmax over Lk=512, 16 threads per row
  {
    const int r = tid >> 4, cdx = tid & 15;
    float m = NEG_BIG;
    for (int j = cdx; j < 512; j += 16) m = fmaxf(m, scores[r * 512 + j]);
    red[r * 16 + cdx] = m;
    __syncthreads();
    if (cdx == 0) {
      float mm = NEG_BIG;
      for (int i = 0; i < 16; ++i) mm = fmaxf(mm, red[r * 16 + i]);
      rowstat[r] = mm;
    }
    __syncthreads();
    const float rmax = rowstat[r];
    float s = 0.0f;
    for (int j = cdx; j < 512; j += 16) {
      float e = __expf(scores[r * 512 + j] - rmax);
      scores[r * 512 + j] = e;
      s += e;
    }
    red[r * 16 + cdx] = s;
    __syncthreads();
    if (cdx == 0) {
      float ss = 0.0f;
      for (int i = 0; i < 16; ++i) ss += red[r * 16 + i];
      rowstat[r] = ss;
    }
    __syncthreads();
    const float inv = __builtin_amdgcn_rcpf(rowstat[r]);
    for (int j = cdx; j < 512; j += 16)
      attn_h[r * 512 + j] = (_Float16)(scores[r * 512 + j] * inv);
  }
  __syncthreads();

  // O[16x128] = attn[16x512] @ V[512x128] : 16 WMMAs per wave
  const int n0 = wave * 16;
  {
    const _Float16* arow = attn_h + (lane & 15) * 512;
    v8f c = {};
#pragma unroll
    for (int kc = 0; kc < 16; ++kc) {
      const int k0  = kc * 32;
      const int kLo = k0 + (hi ? 8 : 0);
      const int kHi = k0 + 16 + (hi ? 8 : 0);
      V16 a;
      a.h[0] = *(const half8*)(arow + kLo);
      a.h[1] = *(const half8*)(arow + kHi);
      const _Float16* vrow = Vh + (b * 512 + kLo) * 128 + n0 + (lane & 15);
      V16 bb;
#pragma unroll
      for (int s = 0; s < 8; ++s) {
        bb.v[s]     = vrow[s * 128];
        bb.v[8 + s] = vrow[(16 + s) * 128];
      }
      c = __builtin_amdgcn_wmma_f32_16x16x32_f16(false, a.v, false, bb.v,
                                                 (short)0, c, false, false);
    }
#pragma unroll
    for (int r = 0; r < 8; ++r) {
      const int row = hi ? 8 + r : r;
      o_lds[row * 128 + n0 + (lane & 15)] = (_Float16)c[r];
    }
  }
  __syncthreads();

  // out[16x128] = O[16x128] @ Wo[128x128] : 4 WMMAs per wave
  {
    const _Float16* arow = o_lds + (lane & 15) * 128;
    v8f c2 = {};
#pragma unroll
    for (int kc = 0; kc < 4; ++kc) {
      const int k0  = kc * 32;
      const int kLo = k0 + (hi ? 8 : 0);
      const int kHi = k0 + 16 + (hi ? 8 : 0);
      V16 a;
      a.h[0] = *(const half8*)(arow + kLo);
      a.h[1] = *(const half8*)(arow + kHi);
      const _Float16* wrow = Woh + kLo * 128 + n0 + (lane & 15);
      V16 bb;
#pragma unroll
      for (int s = 0; s < 8; ++s) {
        bb.v[s]     = wrow[s * 128];
        bb.v[8 + s] = wrow[(16 + s) * 128];
      }
      c2 = __builtin_amdgcn_wmma_f32_16x16x32_f16(false, a.v, false, bb.v,
                                                  (short)0, c2, false, false);
    }
#pragma unroll
    for (int r = 0; r < 8; ++r) {
      const int row = hi ? 8 + r : r;
      out[(b * 512 + qt * 16 + row) * 128 + n0 + (lane & 15)] = c2[r];
    }
  }
}

// ---------------------------------------------------------------- launch
extern "C" void kernel_launch(void* const* d_in, const int* in_sizes, int n_in,
                              void* d_out, int out_size, void* d_ws, size_t ws_size,
                              hipStream_t stream) {
  const float* queries = (const float*)d_in[0];  // (4,512,128)
  const float* keys    = (const float*)d_in[1];  // (4,512,128)
  const float* values  = (const float*)d_in[2];  // (4,512,128)
  const int*   vsl     = (const int*)d_in[3];    // (4,)
  const float* Wq      = (const float*)d_in[4];  // (128,128)
  const float* Wk      = (const float*)d_in[5];  // (128,128)
  const float* wv      = (const float*)d_in[6];  // (128,)
  const float* Wo      = (const float*)d_in[7];  // (128,128)
  float* out = (float*)d_out;

  const int ROWS = 4 * 512;           // 2048
  const int NBIG = ROWS * 128;        // 262144
  const int NW   = 128 * 128;         // 16384

  char* ws = (char*)d_ws;
  _Float16* qh  = (_Float16*)ws; ws += (size_t)NBIG * 2;
  _Float16* kh  = (_Float16*)ws; ws += (size_t)NBIG * 2;
  _Float16* vh  = (_Float16*)ws; ws += (size_t)NBIG * 2;
  _Float16* Wqh = (_Float16*)ws; ws += (size_t)NW * 2;
  _Float16* Wkh = (_Float16*)ws; ws += (size_t)NW * 2;
  _Float16* Woh = (_Float16*)ws; ws += (size_t)NW * 2;
  float*    Qp  = (float*)ws;    ws += (size_t)NBIG * 4;
  float*    Kp  = (float*)ws;    ws += (size_t)NBIG * 4;

  conv_f32_to_f16<<<(NBIG + 255) / 256, 256, 0, stream>>>(queries, qh, NBIG);
  conv_f32_to_f16<<<(NBIG + 255) / 256, 256, 0, stream>>>(keys,    kh, NBIG);
  conv_f32_to_f16<<<(NBIG + 255) / 256, 256, 0, stream>>>(values,  vh, NBIG);
  conv_f32_to_f16<<<(NW   + 255) / 256, 256, 0, stream>>>(Wq, Wqh, NW);
  conv_f32_to_f16<<<(NW   + 255) / 256, 256, 0, stream>>>(Wk, Wkh, NW);
  conv_f32_to_f16<<<(NW   + 255) / 256, 256, 0, stream>>>(Wo, Woh, NW);

  proj_gemm<<<ROWS / 16, 256, 0, stream>>>(qh, Wqh, Qp);   // Qp = Q@Wq
  proj_gemm<<<ROWS / 16, 256, 0, stream>>>(kh, Wkh, Kp);   // Kp = K@Wk

  attn_kernel<<<dim3(32, 4), 256, 0, stream>>>(Qp, Kp, vh, Woh, wv, vsl, out);
}
